// CrossAttention3D_23295902614250
// MI455X (gfx1250) — matrix operand order repaired and checked by reading, not compiled, not run on hardware
//
#include <hip/hip_runtime.h>

// CDNA5 / gfx1250 fused cross-attention.
//   proj_kernel: Q = qin*Wq+bq, K = kv*Wk+bk, V = kv*Wv+bv  (f32 -> f16),
//                V stored transposed [128, Nkv] for B-matrix-friendly reads.
//   attn_kernel: flash-attention, one wave32 per 16-query tile; per-block
//                K/V chunks staged into LDS with GLOBAL_LOAD_ASYNC_TO_LDS_B128
//                (double-buffered, ASYNCcnt), matmuls via
//                v_wmma_f32_16x16x32_f16, B-matrices read from LDS.

typedef __attribute__((ext_vector_type(16))) _Float16 v16h;
typedef __attribute__((ext_vector_type(8)))  _Float16 v8h;
typedef __attribute__((ext_vector_type(8)))  float    v8f;

static constexpr int kB   = 4;
static constexpr int kNQ  = 4096;
static constexpr int kNKV = 4096;
static constexpr int kC   = 128;   // input channels
static constexpr int kF   = 128;   // filters / head dim

__device__ __forceinline__ v16h cat16(v8h a, v8h b) {
  return __builtin_shufflevector(a, b, 0,1,2,3,4,5,6,7,8,9,10,11,12,13,14,15);
}
// B-matrix (32x16 f16): lane holds col N=l&15, element i <-> k = i + 16*hi.
// 16 contiguous halves at p (16B aligned).
__device__ __forceinline__ v16h load_b16(const _Float16* p) {
  return cat16(*(const v8h*)p, *(const v8h*)(p + 8));
}
// A-matrix (16x32 f16): lane holds row M=l&15, element i <-> K=hi*16+i.
// p points at K=hi*16 within the row.
__device__ __forceinline__ v16h load_a16(const _Float16* p) {
  return cat16(*(const v8h*)p, *(const v8h*)(p + 8));
}
__device__ __forceinline__ v8f wmma_ff16(v16h a, v16h b, v8f c) {
  return __builtin_amdgcn_wmma_f32_16x16x32_f16(false, a, false, b, (short)0, c,
                                                false, false);
}

// Async global -> LDS copy, 16 bytes per lane.  VDST VGPR carries the LDS
// byte address (low 32 bits of the generic shared pointer), VADDR the 64-bit
// global address.  Tracked by ASYNCcnt.
__device__ __forceinline__ void async_b128(const _Float16* g, _Float16* l) {
  const unsigned lds = (unsigned)(size_t)(void*)l;
  asm volatile("global_load_async_to_lds_b128 %0, %1, off"
               :: "v"(lds), "v"((unsigned long long)(size_t)g)
               : "memory");
}
__device__ __forceinline__ void wait_async0() {
#if __has_builtin(__builtin_amdgcn_s_wait_asynccnt)
  __builtin_amdgcn_s_wait_asynccnt(0);
#else
  asm volatile("s_wait_asynccnt 0x0" ::: "memory");
#endif
}

// ---------------------------------------------------------------------------
// Kernel 1: projections.  One wave computes one 16x16 output tile.
// 8192 tiles per projection * 3 projections = 24576 waves = 3072 blocks of 256.
// ---------------------------------------------------------------------------
__global__ void __launch_bounds__(256) proj_kernel(
    const float* __restrict__ qin, const float* __restrict__ kvin,
    const float* __restrict__ Wq, const float* __restrict__ bq,
    const float* __restrict__ Wk, const float* __restrict__ bk,
    const float* __restrict__ Wv, const float* __restrict__ bv,
    _Float16* __restrict__ Qh, _Float16* __restrict__ Kh,
    _Float16* __restrict__ Vt) {
  const int lane = threadIdx.x & 31;
  const int wave = threadIdx.x >> 5;
  const int g    = blockIdx.x * 8 + wave;
  const int proj = g >> 13;          // 0=Q, 1=K, 2=V  (8192 tiles each)
  const int t    = g & 8191;
  const int rowbase = (t >> 3) * 16; // 0..16368 over flattened B*N rows
  const int f0      = (t & 7) * 16;  // output feature tile
  const int hi = lane >> 4;
  const int ln = lane & 15;

  const float* X    = (proj == 0) ? qin : kvin;
  const float* W    = (proj == 0) ? Wq : (proj == 1) ? Wk : Wv;
  const float* bias = (proj == 0) ? bq : (proj == 1) ? bk : bv;

  const float* xrow = X + (size_t)(rowbase + ln) * kC;
  const int    fcol = f0 + ln;

  v8f acc = {};
#pragma unroll
  for (int kk = 0; kk < 4; ++kk) {
    // A: row (rowbase+ln), contiguous-k layout: k = i + 16*hi
    const int c0 = kk * 32 + hi * 16;
    v16h a;
#pragma unroll
    for (int e = 0; e < 8; ++e) {
      a[e]     = (_Float16)xrow[c0 + e];
      a[e + 8] = (_Float16)xrow[c0 + 8 + e];
    }
    // B: column fcol of W, contiguous-k layout: k = i + 16*hi
    v16h bm;
#pragma unroll
    for (int i = 0; i < 16; ++i) {
      const int c = kk * 32 + 16 * hi + i;
      bm[i] = (_Float16)W[(size_t)c * kF + fcol];
    }
    acc = wmma_ff16(a, bm, acc);
  }
  const float bb = bias[fcol];
#pragma unroll
  for (int j = 0; j < 8; ++j) acc[j] += bb;

  if (proj < 2) {
    _Float16* dst = (proj == 0) ? Qh : Kh;
#pragma unroll
    for (int j = 0; j < 8; ++j) {
      const int m = rowbase + j + hi * 8;        // C/D layout row
      dst[(size_t)m * kF + fcol] = (_Float16)acc[j];
    }
  } else {
    // V transposed: Vt[b][f][key]; VGPR j holds 8 consecutive keys -> b128 store
    const int b       = rowbase >> 12;           // /4096
    const int keybase = (rowbase & (kNKV - 1)) + hi * 8;
    v8h pk;
#pragma unroll
    for (int j = 0; j < 8; ++j) pk[j] = (_Float16)acc[j];
    *(v8h*)(Vt + (size_t)(b * kF + fcol) * kNKV + keybase) = pk;
  }
}

// ---------------------------------------------------------------------------
// Kernel 2: flash attention.  One wave per 16-query tile, 8 waves per block
// sharing async-staged K/V chunks.  4*256 tiles = 1024 waves = 128 blocks.
// ---------------------------------------------------------------------------
// Stage one 32-key chunk: K rows are contiguous (8 KB), V rows are 64 B each.
// 256 threads x 2 b128 per array.
__device__ __forceinline__ void stage_chunk(const _Float16* kbat,
                                            const _Float16* vbat, int kb,
                                            _Float16* ldsK, _Float16* ldsV,
                                            int tid) {
  const _Float16* ksrc = kbat + (size_t)kb * kC;     // 8 KB contiguous
  async_b128(ksrc + (size_t)tid * 8, ldsK + tid * 8);
  async_b128(ksrc + (size_t)(tid + 256) * 8, ldsK + (tid + 256) * 8);
  // V: seg s -> feat = s>>2, 16B part = s&3 ; LDS layout [feat][32 keys]
  const int s0 = tid, s1 = tid + 256;
  async_b128(vbat + (size_t)(s0 >> 2) * kNKV + kb + (s0 & 3) * 8, ldsV + s0 * 8);
  async_b128(vbat + (size_t)(s1 >> 2) * kNKV + kb + (s1 & 3) * 8, ldsV + s1 * 8);
}

__global__ void __launch_bounds__(256) attn_kernel(
    const _Float16* __restrict__ Qh, const _Float16* __restrict__ Kh,
    const _Float16* __restrict__ Vt, float* __restrict__ out) {
  __shared__ _Float16 sK[2 * 32 * kC];    // 2 x 8 KB  K chunk  [key][c]
  __shared__ _Float16 sV[2 * kF * 32];    // 2 x 8 KB  V chunk  [feat][key]
  __shared__ _Float16 sP[8 * 16 * 32];    // 8 KB      per-wave P staging

  const int tid   = threadIdx.x;
  const int lane  = tid & 31;
  const int wave  = tid >> 5;
  const int g     = blockIdx.x * 8 + wave;
  const int b     = g >> 8;                 // batch
  const int qbase = (g & 255) * 16;         // query tile base
  const int hi = lane >> 4;
  const int ln = lane & 15;
  _Float16* pbuf = sP + wave * (16 * 32);

  // Q tile as 4 A-matrices (K = C = 128), pre-scaled by 1/sqrt(F)
  const _Float16* qrow = Qh + (size_t)(b * kNQ + qbase + ln) * kC;
  const _Float16 qscale = (_Float16)0.08838834764831845f;  // 1/sqrt(128)
  v16h aQ[4];
#pragma unroll
  for (int kk = 0; kk < 4; ++kk)
    aQ[kk] = load_a16(qrow + kk * 32 + hi * 16) * qscale;

  float rowmax[8], rowsum[8];
#pragma unroll
  for (int j = 0; j < 8; ++j) { rowmax[j] = -1e30f; rowsum[j] = 0.0f; }
  v8f O[8];
  const v8f vzero = {};
#pragma unroll
  for (int ft = 0; ft < 8; ++ft) O[ft] = vzero;

  const _Float16* kbat = Kh + (size_t)b * kNKV * kC;
  const _Float16* vbat = Vt + (size_t)b * kF * kNKV;

  const int niter = kNKV / 32;
  stage_chunk(kbat, vbat, 0, sK, sV, tid);               // prologue -> buf 0

  for (int it = 0; it < niter; ++it) {
    const int cur = it & 1;
    wait_async0();       // this wave's staged chunk landed in LDS
    __syncthreads();     // all waves' chunks visible; prior-buffer reads done
    if (it + 1 < niter)  // overlap next chunk's DMA with this chunk's compute
      stage_chunk(kbat, vbat, (it + 1) * 32,
                  sK + (1 - cur) * (32 * kC), sV + (1 - cur) * (kF * 32), tid);

    const _Float16* Kt = sK + cur * (32 * kC);
    const _Float16* Vtile = sV + cur * (kF * 32);

    // ---- S = Q * K^T for 32 keys (two 16x16 C/D tiles), B from LDS ----
    v8f S0 = vzero, S1 = vzero;
#pragma unroll
    for (int kk = 0; kk < 4; ++kk) {
      v16h bK0 = load_b16(Kt + ln * kC + kk * 32 + hi * 16);
      v16h bK1 = load_b16(Kt + (16 + ln) * kC + kk * 32 + hi * 16);
      S0 = wmma_ff16(aQ[kk], bK0, S0);
      S1 = wmma_ff16(aQ[kk], bK1, S1);
    }

    // ---- online softmax (row M=j+8*hi lives across the 16-lane half) ----
    float nm[8], alpha[8];
#pragma unroll
    for (int j = 0; j < 8; ++j) {
      float m = fmaxf(S0[j], S1[j]);
      m = fmaxf(m, __shfl_xor(m, 1));
      m = fmaxf(m, __shfl_xor(m, 2));
      m = fmaxf(m, __shfl_xor(m, 4));
      m = fmaxf(m, __shfl_xor(m, 8));
      nm[j]     = fmaxf(rowmax[j], m);
      alpha[j]  = __expf(rowmax[j] - nm[j]);
      rowmax[j] = nm[j];
    }
    float p0[8], p1[8];
#pragma unroll
    for (int j = 0; j < 8; ++j) {
      p0[j] = __expf(S0[j] - nm[j]);
      p1[j] = __expf(S1[j] - nm[j]);
      float s = p0[j] + p1[j];
      s += __shfl_xor(s, 1);
      s += __shfl_xor(s, 2);
      s += __shfl_xor(s, 4);
      s += __shfl_xor(s, 8);
      rowsum[j] = rowsum[j] * alpha[j] + s;
    }
#pragma unroll
    for (int ft = 0; ft < 8; ++ft)
#pragma unroll
      for (int j = 0; j < 8; ++j) O[ft][j] *= alpha[j];

    // ---- P: C/D layout -> LDS -> A layout (wave-private region) ----
#pragma unroll
    for (int j = 0; j < 8; ++j) {
      const int m = j + hi * 8;
      pbuf[m * 32 + ln]      = (_Float16)p0[j];
      pbuf[m * 32 + 16 + ln] = (_Float16)p1[j];
    }
    __syncthreads();  // uniform; fences the P store -> load turnaround
    v16h aP = load_a16(pbuf + ln * 32 + hi * 16);

    // ---- O += P * V (8 feature tiles), B from LDS ----
#pragma unroll
    for (int ft = 0; ft < 8; ++ft) {
      v16h bV = load_b16(Vtile + (ft * 16 + ln) * 32 + hi * 16);
      O[ft] = wmma_ff16(aP, bV, O[ft]);
    }
  }

  // ---- normalize and write out [B, Nq, F] f32 ----
  float inv[8];
#pragma unroll
  for (int j = 0; j < 8; ++j) inv[j] = 1.0f / rowsum[j];
#pragma unroll
  for (int ft = 0; ft < 8; ++ft)
#pragma unroll
    for (int j = 0; j < 8; ++j) {
      const int m = qbase + j + hi * 8;
      out[(size_t)(b * kNQ + m) * kF + ft * 16 + ln] = O[ft][j] * inv[j];
    }
}

// ---------------------------------------------------------------------------
extern "C" void kernel_launch(void* const* d_in, const int* in_sizes, int n_in,
                              void* d_out, int out_size, void* d_ws,
                              size_t ws_size, hipStream_t stream) {
  const float* qin  = (const float*)d_in[0];
  const float* kvin = (const float*)d_in[1];
  const float* Wq   = (const float*)d_in[2];
  const float* bq   = (const float*)d_in[3];
  const float* Wk   = (const float*)d_in[4];
  const float* bk   = (const float*)d_in[5];
  const float* Wv   = (const float*)d_in[6];
  const float* bv   = (const float*)d_in[7];

  _Float16* Qh = (_Float16*)d_ws;                       // [B,Nq,128]  f16
  _Float16* Kh = Qh + (size_t)kB * kNQ * kF;            // [B,Nkv,128] f16
  _Float16* Vt = Kh + (size_t)kB * kNKV * kF;           // [B,128,Nkv] f16

  proj_kernel<<<3072, 256, 0, stream>>>(qin, kvin, Wq, bq, Wk, bk, Wv, bv,
                                        Qh, Kh, Vt);
  attn_kernel<<<128, 256, 0, stream>>>(Qh, Kh, Vt, (float*)d_out);
}
